// ResUNetWithTexture_2199023256191
// MI455X (gfx1250) — compile-verified, matrix operand order checked by reading, not run on hardware
//
#include <hip/hip_runtime.h>
#include <hip/hip_bf16.h>

// ---------------------------------------------------------------------------
// ResUNetWithTexture for MI455X (gfx1250).
//
// Every conv / deconv is one implicit-GEMM engine on V_WMMA_F32_16X16X32_F16:
//     M = B*OH*OW (pixels), N = Cout, K = Cin*kh*kw
// Workgroup = 256 threads = 8 wave32s as 4(M) x 2(N); each wave register-
// blocks 2x2 16x16 accumulators -> 128x64 tile, K stepped by 32
// (4 WMMA per wave per step). A-tile (fused im2col f32->f16) staged to LDS
// by VALU; B-tile (pre-packed f16 weights) DMA'd into LDS by the Tensor Data
// Mover (tensor_load_to_lds + s_wait_tensorcnt), with TDM padding producing
// the bank-conflict-avoiding 40-half LDS row pitch.
// Epilogue fuses conv-bias, BN affine, residual-add, ReLU.
// ---------------------------------------------------------------------------

typedef __attribute__((ext_vector_type(16))) _Float16 v16h;
typedef __attribute__((ext_vector_type(8)))  _Float16 v8h;
typedef __attribute__((ext_vector_type(8)))  float    v8f;
typedef unsigned int u32x4 __attribute__((ext_vector_type(4)));
typedef int          i32x8 __attribute__((ext_vector_type(8)));
typedef int          i32x4 __attribute__((ext_vector_type(4)));

#ifndef __has_builtin
#define __has_builtin(x) 0
#endif
#if defined(__gfx1250__) && __has_builtin(__builtin_amdgcn_tensor_load_to_lds)
#define HAVE_TDM 1
#else
#define HAVE_TDM 0
#endif

struct ConvDesc {
    int B, Cin, H, W, Cout, OH, OW, kh, kw, stride, pad;
    int OHfull, OWfull, oscale, orow, ocol;   // strided output placement (deconv)
    int relu;
};

// ------------------------------ weight packing -----------------------------

__global__ void pack_f32_to_f16_k(const float* __restrict__ src,
                                  _Float16* __restrict__ dst, int n) {
    int i = blockIdx.x * blockDim.x + threadIdx.x;
    if (i < n) dst[i] = (_Float16)src[i];
}

// conv_transpose weights w[ci][co][r][c] (IOHW 2x2) -> 4 slabs [rc][co][ci]
__global__ void pack_deconv_w_k(const float* __restrict__ src,
                                _Float16* __restrict__ dst, int Cin, int Cout) {
    int i = blockIdx.x * blockDim.x + threadIdx.x;
    int total = Cin * Cout * 4;
    if (i >= total) return;
    int ci = i % Cin; int t = i / Cin; int co = t % Cout; int rc = t / Cout;
    int r = rc >> 1, c = rc & 1;
    dst[i] = (_Float16)src[((ci * Cout + co) * 2 + r) * 2 + c];
}

// --------------------------- WMMA implicit-GEMM conv -----------------------

__global__ __launch_bounds__(256)
void conv_wmma_k(const float* __restrict__ x, const _Float16* __restrict__ wpk,
                 const float* __restrict__ cbias, const float* __restrict__ gam,
                 const float* __restrict__ bet, const float* __restrict__ resid,
                 float* __restrict__ out, ConvDesc dd) {
    __shared__ __align__(16) _Float16 lA[128][40];  // 128 pixels x 32 K (+8 pad)
    __shared__ __align__(16) _Float16 lB[64][40];   // 64 couts  x 32 K (+8 pad)

    const int tid  = threadIdx.x;
    const int lane = tid & 31, wave = tid >> 5;
    const int wm = wave & 3, wn = wave >> 2;        // wave tile coords (32x32 each)
    const int l16 = lane & 15, hl = lane >> 4;      // ISA 16-lane half split
    const int OHW  = dd.OH * dd.OW;
    const int Mtot = dd.B * OHW;
    const int Ktot = dd.Cin * dd.kh * dd.kw;
    const int khkw = dd.kh * dd.kw;
    const int mBase  = blockIdx.x * 128;
    const int coBase = blockIdx.y * 64;

    // A-staging assignment: row amr (0..127), K-half akg (0/1) -> 16 K each.
    const int amr = tid & 127;
    const int akg = tid >> 7;
    const int am  = mBase + amr;
    const bool amok = am < Mtot;
    int ab = 0, ihB = 0, iwB = 0;
    if (amok) {
        ab = am / OHW; int r = am % OHW;
        ihB = (r / dd.OW) * dd.stride - dd.pad;
        iwB = (r % dd.OW) * dd.stride - dd.pad;
    }
    const float* xb = x + (size_t)ab * dd.Cin * dd.H * dd.W;

    v8f a00 = {}, a01 = {}, a10 = {}, a11 = {};
    union Frag { v16h v; v8h h[2]; };

    for (int k0 = 0; k0 < Ktot; k0 += 32) {
#if HAVE_TDM
        if (wave == 0) {
            // D# for a [64 co][32 k] f16 tile of W[Cout][Ktot]; LDS rows padded
            // by TDM: pad_interval 16 DWORDs (enc 3), pad_amount 4 DWORDs
            // (enc 3) -> 40-half row pitch matching lB.
            unsigned long long gaddr =
                (unsigned long long)(const void*)wpk +
                (((unsigned long long)(unsigned)coBase * (unsigned)Ktot +
                  (unsigned)k0) << 1);
            unsigned ldsb = (unsigned)(unsigned long long)(const void*)&lB[0][0];
            u32x4 g0;
            g0[0] = 1u;                                        // count=1, user D#
            g0[1] = ldsb;                                      // lds_addr
            g0[2] = (unsigned)gaddr;                           // global_addr lo
            g0[3] = ((unsigned)(gaddr >> 32) & 0x01FFFFFFu)    // global_addr hi
                    | (2u << 30);                              // type = image
            i32x8 g1;
            g1[0] = (int)((1u << 16)      // data_size = 2B (f16)
                          | (1u << 20)    // pad_enable
                          | (3u << 22)    // pad_interval: 16 DWORDs
                          | (3u << 25));  // pad_amount:   4 DWORDs
            g1[1] = (int)(((unsigned)Ktot & 0xFFFFu) << 16);   // tensor_dim0 lo16
            g1[2] = (int)((((unsigned)Ktot >> 16) & 0xFFFFu)   // tensor_dim0 hi16
                          | (((unsigned)dd.Cout & 0xFFFFu) << 16)); // dim1 lo16
            g1[3] = (int)((((unsigned)dd.Cout >> 16) & 0xFFFFu)
                          | (32u << 16));                      // tile_dim0 = 32
            g1[4] = 64;                                        // tile_dim1 = 64
            g1[5] = Ktot;                                      // dim0_stride lo32
            g1[6] = 0;
            g1[7] = 0;
            i32x4 z4 = {0, 0, 0, 0};
#if __clang_major__ >= 23
            i32x8 z8 = {0, 0, 0, 0, 0, 0, 0, 0};
            __builtin_amdgcn_tensor_load_to_lds(g0, g1, z4, z4, z8, 0);
#else
            __builtin_amdgcn_tensor_load_to_lds(g0, g1, z4, z4, 0);
#endif
        }
#endif
        // ---- stage A: fused im2col + f32->f16, incremental (ci,kr,kc) ----
        {
            int ks = k0 + akg * 16;
            int ci = ks / khkw; int rem = ks - ci * khkw;
            int kr = rem / dd.kw; int kc = rem - kr * dd.kw;
            #pragma unroll
            for (int u = 0; u < 16; ++u) {
                float v = 0.f;
                if (amok && (ks + u) < Ktot) {
                    int ih = ihB + kr, iw = iwB + kc;
                    if (ih >= 0 && ih < dd.H && iw >= 0 && iw < dd.W)
                        v = xb[((size_t)ci * dd.H + ih) * dd.W + iw];
                }
                lA[amr][akg * 16 + u] = (_Float16)v;
                if (++kc == dd.kw) { kc = 0; if (++kr == dd.kh) { kr = 0; ++ci; } }
            }
            if (amok && k0 + 32 < Ktot) {               // global_prefetch_b8 path
                int ci2 = (k0 + 32) / khkw;
                int ph = ihB < 0 ? 0 : ihB, pw = iwB < 0 ? 0 : iwB;
                __builtin_prefetch(&xb[((size_t)ci2 * dd.H + ph) * dd.W + pw], 0, 1);
            }
        }
#if HAVE_TDM
        if (wave == 0) __builtin_amdgcn_s_wait_tensorcnt(0);
#else
        {   // ---- manual B staging fallback (no TDM builtin) ----
            const int bco = tid & 63, bkg = tid >> 6;
            const int gco = coBase + bco;
            #pragma unroll
            for (int u = 0; u < 8; ++u) {
                int kl = bkg * 8 + u;
                int k  = k0 + kl;
                _Float16 v = (_Float16)0.f;
                if (gco < dd.Cout && k < Ktot) v = wpk[(size_t)gco * Ktot + k];
                lB[bco][kl] = v;
            }
        }
#endif
        __syncthreads();

        // ---- fragments per ISA 16-bit A/B layout + 2x2 WMMA block ----
        Frag fa0, fa1, fb0, fb1;
        const int kb = hl * 8;                          // lanes>=16 start at K=8
        const int ar = wm * 32 + l16;
        fa0.h[0] = *(const v8h*)&lA[ar][kb];
        fa0.h[1] = *(const v8h*)&lA[ar][kb + 16];
        fa1.h[0] = *(const v8h*)&lA[ar + 16][kb];
        fa1.h[1] = *(const v8h*)&lA[ar + 16][kb + 16];
        const int br = wn * 32 + l16;
        fb0.h[0] = *(const v8h*)&lB[br][kb];
        fb0.h[1] = *(const v8h*)&lB[br][kb + 16];
        fb1.h[0] = *(const v8h*)&lB[br + 16][kb];
        fb1.h[1] = *(const v8h*)&lB[br + 16][kb + 16];
        a00 = __builtin_amdgcn_wmma_f32_16x16x32_f16(false, fa0.v, false, fb0.v,
                                                     (short)0, a00, false, false);
        a01 = __builtin_amdgcn_wmma_f32_16x16x32_f16(false, fa0.v, false, fb1.v,
                                                     (short)0, a01, false, false);
        a10 = __builtin_amdgcn_wmma_f32_16x16x32_f16(false, fa1.v, false, fb0.v,
                                                     (short)0, a10, false, false);
        a11 = __builtin_amdgcn_wmma_f32_16x16x32_f16(false, fa1.v, false, fb1.v,
                                                     (short)0, a11, false, false);
        __syncthreads();
    }

    // ---- fused epilogue: (acc + conv_bias)*gamma + beta [+res] [relu] ----
    v8f accs[2][2] = {{a00, a01}, {a10, a11}};
    #pragma unroll
    for (int ms = 0; ms < 2; ++ms) {
        #pragma unroll
        for (int ns = 0; ns < 2; ++ns) {
            const int co = coBase + wn * 32 + ns * 16 + l16;  // C/D col = Cout
            if (co >= dd.Cout) continue;
            const float g  = gam   ? gam[co]   : 1.f;
            const float bb = bet   ? bet[co]   : 0.f;
            const float cb = cbias ? cbias[co] : 0.f;
            v8f acc = accs[ms][ns];
            #pragma unroll
            for (int r = 0; r < 8; ++r) {                     // VGPR r -> pixel
                int m = mBase + wm * 32 + ms * 16 + r + 8 * hl;
                if (m >= Mtot) continue;
                int b = m / OHW; int rem = m % OHW;
                int oh = rem / dd.OW, ow = rem % dd.OW;
                size_t oidx = (((size_t)b * dd.Cout + co) * dd.OHfull
                               + (oh * dd.oscale + dd.orow)) * dd.OWfull
                              + (ow * dd.oscale + dd.ocol);
                float v = (acc[r] + cb) * g + bb;
                if (resid) v += resid[oidx];
                if (dd.relu) v = fmaxf(v, 0.f);
                out[oidx] = v;
            }
        }
    }
}

// --------------------------- texture pyramid (VALU) ------------------------

__global__ void texture_k(const float* __restrict__ x, float* __restrict__ out,
                          int B, int C, int H, int W, int d) {
    int i = blockIdx.x * blockDim.x + threadIdx.x;
    int HW = H * W;
    if (i >= B * C * HW) return;
    int w0 = i % W; int t = i / W; int h0 = t % H; t /= H;
    int c = t % C; int b = t / C;
    int k = 3 + 2 * (d - 1);
    int K = k * k;
    int p = d * (k - 1) / 2;
    const float* xc = x + ((size_t)(b * C + c)) * HW;

    float sum = 0.f, sumsq = 0.f, splogp = 0.f;
    for (int r = 0; r < k; ++r) {
        int ih = h0 - p + r * d;
        for (int cc = 0; cc < k; ++cc) {
            int iw = w0 - p + cc * d;
            float v = (ih >= 0 && ih < H && iw >= 0 && iw < W) ? xc[ih * W + iw] : 0.f;
            sum += v; sumsq += v * v; splogp += v * __logf(v + 1e-6f);
        }
    }
    float mean   = sum / K;
    float scent2 = fmaxf(sumsq - K * mean * mean, 0.f);
    float stdv   = sqrtf(scent2 / (K - 1)) + 1e-6f;
    float contrast = scent2 / (K * stdv * stdv);
    float energy   = sumsq / K;
    float entropy  = -splogp / K;
    float homog = 0.f;
    for (int r = 0; r < k; ++r) {
        int ih = h0 - p + r * d;
        for (int cc = 0; cc < k; ++cc) {
            int iw = w0 - p + cc * d;
            float v = (ih >= 0 && ih < H && iw >= 0 && iw < W) ? xc[ih * W + iw] : 0.f;
            homog += 1.f / (1.f + fabsf(v - mean));
        }
    }
    homog /= K;
    float f[4] = {contrast, energy, entropy, homog};
    #pragma unroll
    for (int j = 0; j < 4; ++j) {
        // THETA = 1: M = exp(log(f+eps) - 0.5)
        float M = __expf(__logf(f[j] + 1e-6f) - 0.5f);
        out[(((size_t)b * (C * 4) + c * 4 + j) * H + h0) * W + w0] = M;
    }
}

// ------------------------------- helpers -----------------------------------

__global__ void maxpool_k(const float* __restrict__ in, float* __restrict__ out,
                          int B, int C, int H, int W, int OH, int OW) {
    size_t i = (size_t)blockIdx.x * blockDim.x + threadIdx.x;
    size_t tot = (size_t)B * C * OH * OW;
    if (i >= tot) return;
    int ow = i % OW; size_t t = i / OW; int oh = t % OH; t /= OH;
    int c = t % C; int b = (int)(t / C);
    float m = -3.4e38f;
    for (int r = 0; r < 3; ++r) {
        int ih = oh * 2 - 1 + r;
        if (ih < 0 || ih >= H) continue;
        for (int cc = 0; cc < 3; ++cc) {
            int iw = ow * 2 - 1 + cc;
            if (iw < 0 || iw >= W) continue;
            m = fmaxf(m, in[(((size_t)b * C + c) * H + ih) * W + iw]);
        }
    }
    out[i] = m;
}

__global__ void copy_ch_k(const float* __restrict__ src, float* __restrict__ dst,
                          int B, int Csrc, int HW, int Cdst, int coff) {
    size_t i = (size_t)blockIdx.x * blockDim.x + threadIdx.x;
    size_t tot = (size_t)B * Csrc * HW;
    if (i >= tot) return;
    int hw = i % HW; size_t t = i / HW; int c = t % Csrc; int b = (int)(t / Csrc);
    dst[((size_t)b * Cdst + coff + c) * HW + hw] = src[i];
}

__global__ void gap_k(const float* __restrict__ in, float* __restrict__ out, int HW) {
    __shared__ float sm[256];
    int bc = blockIdx.x;
    float a = 0.f;
    for (int i = threadIdx.x; i < HW; i += 256) a += in[(size_t)bc * HW + i];
    sm[threadIdx.x] = a;
    __syncthreads();
    for (int s = 128; s > 0; s >>= 1) {
        if (threadIdx.x < s) sm[threadIdx.x] += sm[threadIdx.x + s];
        __syncthreads();
    }
    if (threadIdx.x == 0) out[bc] = sm[0] / HW;
}

__global__ void fc_k(const float* __restrict__ in, const float* __restrict__ w,
                     const float* __restrict__ bias, float* __restrict__ out,
                     int B, int Cin, int Cout, int relu, int sig) {
    int i = blockIdx.x * blockDim.x + threadIdx.x;
    if (i >= B * Cout) return;
    int o = i % Cout, b = i / Cout;
    float a = bias ? bias[o] : 0.f;
    for (int c = 0; c < Cin; ++c) a += in[b * Cin + c] * w[o * Cin + c];
    if (relu) a = fmaxf(a, 0.f);
    if (sig) a = 1.f / (1.f + __expf(-a));
    out[i] = a;
}

__global__ void softmax4_k(const float* __restrict__ in, float* __restrict__ out, int B) {
    int b = blockIdx.x * blockDim.x + threadIdx.x;
    if (b >= B) return;
    float m = in[b * 4];
    for (int i = 1; i < 4; ++i) m = fmaxf(m, in[b * 4 + i]);
    float e[4], s = 0.f;
    for (int i = 0; i < 4; ++i) { e[i] = __expf(in[b * 4 + i] - m); s += e[i]; }
    for (int i = 0; i < 4; ++i) out[b * 4 + i] = e[i] / s;
}

__global__ void fuse_k(const float* __restrict__ s0, const float* __restrict__ s1,
                       const float* __restrict__ s2, const float* __restrict__ s3,
                       const float* __restrict__ wsel, float* __restrict__ out,
                       int B, int C, int HW) {
    size_t i = (size_t)blockIdx.x * blockDim.x + threadIdx.x;
    size_t tot = (size_t)B * C * HW;
    if (i >= tot) return;
    int b = (int)(i / ((size_t)C * HW));
    const float* wv = wsel + b * 4;
    out[i] = s0[i] * wv[0] + s1[i] * wv[1] + s2[i] * wv[2] + s3[i] * wv[3];
}

__global__ void scale_ch_k(float* __restrict__ buf, const float* __restrict__ e,
                           int B, int C, int HW) {
    size_t i = (size_t)blockIdx.x * blockDim.x + threadIdx.x;
    size_t tot = (size_t)B * C * HW;
    if (i >= tot) return;
    int c = (int)((i / HW) % C);
    int b = (int)(i / ((size_t)C * HW));
    buf[i] *= e[b * C + c];
}

__global__ void add3_k(const float* __restrict__ x, const float* __restrict__ tex64,
                       float* __restrict__ out, int B, int HW) {
    size_t i = (size_t)blockIdx.x * blockDim.x + threadIdx.x;
    size_t tot = (size_t)B * 3 * HW;
    if (i >= tot) return;
    int hw = i % HW; int c = (int)((i / HW) % 3); int b = (int)(i / ((size_t)3 * HW));
    out[i] = x[i] + tex64[((size_t)b * 64 + c) * HW + hw];
}

// --------------------------------- driver ----------------------------------
//
// Input flattening assumption: d_in[0] = x, then params leaves in
// jax tree order (nested dicts flattened with sorted keys, lists in order):
//   dec(20 leaves), res.layers(156), res.stem(3: b,g,w), tex(42).
// Workspace is carved with a bump allocator (assumed large enough).

extern "C" void kernel_launch(void* const* d_in, const int* in_sizes, int n_in,
                              void* d_out, int out_size, void* d_ws, size_t ws_size,
                              hipStream_t stream) {
    const int H0 = 256, W0 = 256, HW = H0 * W0;
    const int B = in_sizes[0] / (3 * HW);

    auto F = [&](int i) -> const float* { return (const float*)d_in[i]; };

    // ---- workspace carving ----
    size_t cur = 0;
    auto alloc_f = [&](size_t n) -> float* {
        float* p = (float*)((char*)d_ws + cur);
        cur += (n * sizeof(float) + 255) & ~(size_t)255;
        return p;
    };
    auto alloc_h = [&](size_t n) -> _Float16* {
        _Float16* p = (_Float16*)((char*)d_ws + cur);
        cur += (n * sizeof(_Float16) + 255) & ~(size_t)255;
        return p;
    };

    float* tpyr   = alloc_f((size_t)4 * B * 12 * HW);
    float* bin    = alloc_f((size_t)B * 15 * HW);
    float* ttex   = alloc_f((size_t)B * 64 * HW);
    float* bro[4];
    for (int j = 0; j < 4; ++j) bro[j] = alloc_f((size_t)B * 64 * HW);
    float* fusedT = alloc_f((size_t)B * 64 * HW);
    float* fused  = alloc_f((size_t)B * 64 * HW);
    float* s0     = alloc_f((size_t)B * 64);
    float* a1     = alloc_f((size_t)B * 16);
    float* logit  = alloc_f((size_t)B * 4);
    float* wsel   = alloc_f((size_t)B * 4);
    float* se1    = alloc_f((size_t)B * 4);
    float* evec   = alloc_f((size_t)B * 64);
    float* xin    = alloc_f((size_t)B * 3 * HW);
    float* stemo  = alloc_f((size_t)B * 64 * 128 * 128);
    float* pool   = alloc_f((size_t)B * 64 * 64 * 64);
    float* xf1    = alloc_f((size_t)B * 256 * 64 * 64);
    float* xf2    = alloc_f((size_t)B * 512 * 32 * 32);
    float* xf3    = alloc_f((size_t)B * 1024 * 16 * 16);
    float* xf4    = alloc_f((size_t)B * 2048 * 8 * 8);
    float* bufA   = alloc_f((size_t)B * 256 * 64 * 64);
    float* bufB   = alloc_f((size_t)B * 256 * 64 * 64);
    float* T1     = alloc_f((size_t)B * 128 * 64 * 64);
    float* T2     = alloc_f((size_t)B * 128 * 64 * 64);
    float* SC     = alloc_f((size_t)B * 256 * 64 * 64);
    float* D1     = alloc_f((size_t)B * 256 * 64 * 64);
    float* D2     = alloc_f((size_t)B * 256 * 64 * 64);
    float* CAT    = alloc_f((size_t)B * 512 * 64 * 64);
    _Float16* wpack = alloc_h((size_t)19 * 1024 * 1024);  // >= largest K*Cout

    // ---- host-side conv launchers (stream-ordered, so wpack is reusable) ----
    auto conv = [&](const float* xp, const float* w32, const float* cb,
                    const float* g, const float* bt, const float* rs, float* outp,
                    int Bb, int Cin, int Hh, int Ww, int Cout,
                    int kh, int kw, int s, int p, int relu) {
        int OH = (Hh + 2 * p - kh) / s + 1, OW = (Ww + 2 * p - kw) / s + 1;
        int n = Cout * Cin * kh * kw;
        pack_f32_to_f16_k<<<(n + 255) / 256, 256, 0, stream>>>(w32, wpack, n);
        ConvDesc dd{Bb, Cin, Hh, Ww, Cout, OH, OW, kh, kw, s, p,
                    OH, OW, 1, 0, 0, relu};
        dim3 grid((Bb * OH * OW + 127) / 128, (Cout + 63) / 64);
        conv_wmma_k<<<grid, 256, 0, stream>>>(xp, wpack, cb, g, bt, rs, outp, dd);
    };

    auto deconv = [&](const float* xp, const float* w32, const float* cb,
                      float* outp, int Bb, int Cin, int IH, int IW, int Cout) {
        int n = Cin * Cout * 4;
        pack_deconv_w_k<<<(n + 255) / 256, 256, 0, stream>>>(w32, wpack, Cin, Cout);
        for (int rc = 0; rc < 4; ++rc) {       // 4 strided 1x1 GEMMs
            ConvDesc dd{Bb, Cin, IH, IW, Cout, IH, IW, 1, 1, 1, 0,
                        2 * IH, 2 * IW, 2, rc >> 1, rc & 1, 0};
            dim3 grid((Bb * IH * IW + 127) / 128, (Cout + 63) / 64);
            conv_wmma_k<<<grid, 256, 0, stream>>>(
                xp, wpack + (size_t)rc * Cout * Cin, cb,
                nullptr, nullptr, nullptr, outp, dd);
        }
    };

    const float* x = F(0);

    // ================= texture encoder =================
    for (int d = 1; d <= 4; ++d) {
        int tot = B * 3 * HW;
        texture_k<<<(tot + 255) / 256, 256, 0, stream>>>(
            x, tpyr + (size_t)(d - 1) * B * 12 * HW, B, 3, H0, W0, d);
    }
    for (int j = 0; j < 4; ++j) {
        size_t n3 = (size_t)B * 3 * HW, n12 = (size_t)B * 12 * HW;
        copy_ch_k<<<(unsigned)((n3 + 255) / 256), 256, 0, stream>>>(x, bin, B, 3, HW, 15, 0);
        copy_ch_k<<<(unsigned)((n12 + 255) / 256), 256, 0, stream>>>(
            tpyr + (size_t)j * B * 12 * HW, bin, B, 12, HW, 15, 3);
        int tb = 180 + 8 * j;   // b1,b2,cb1,cb2,g1,g2,w1,w2
        conv(bin,  F(tb + 6), F(tb + 2), F(tb + 4), F(tb + 0), nullptr, ttex,
             B, 15, H0, W0, 64, 3, 3, 1, 1, 1);
        conv(ttex, F(tb + 7), F(tb + 3), F(tb + 5), F(tb + 1), nullptr, bro[j],
             B, 64, H0, W0, 64, 3, 3, 1, 1, 1);
    }
    // scale selector: attention from branch 0
    gap_k<<<B * 64, 256, 0, stream>>>(bro[0], s0, HW);
    fc_k<<<(B * 16 + 63) / 64, 64, 0, stream>>>(s0, F(220), F(218), a1, B, 64, 16, 1, 0);
    fc_k<<<(B * 4 + 63) / 64, 64, 0, stream>>>(a1, F(221), F(219), logit, B, 16, 4, 0, 0);
    softmax4_k<<<(B + 63) / 64, 64, 0, stream>>>(logit, wsel, B);
    {
        size_t tot = (size_t)B * 64 * HW;
        fuse_k<<<(unsigned)((tot + 255) / 256), 256, 0, stream>>>(
            bro[0], bro[1], bro[2], bro[3], wsel, fusedT, B, 64, HW);
    }
    conv(fusedT, F(213), F(212), nullptr, nullptr, nullptr, fused,
         B, 64, H0, W0, 64, 1, 1, 1, 0, 0);
    // SE block
    gap_k<<<B * 64, 256, 0, stream>>>(fused, s0, HW);
    fc_k<<<(B * 4 + 63) / 64, 64, 0, stream>>>(s0, F(216), F(214), se1, B, 64, 4, 1, 0);
    fc_k<<<(B * 64 + 63) / 64, 64, 0, stream>>>(se1, F(217), F(215), evec, B, 4, 64, 0, 1);
    {
        size_t tot = (size_t)B * 64 * HW;
        scale_ch_k<<<(unsigned)((tot + 255) / 256), 256, 0, stream>>>(fused, evec, B, 64, HW);
        size_t tot3 = (size_t)B * 3 * HW;
        add3_k<<<(unsigned)((tot3 + 255) / 256), 256, 0, stream>>>(x, fused, xin, B, HW);
    }

    // ================= ResNet-50 encoder =================
    conv(xin, F(179), nullptr, F(178), F(177), nullptr, stemo,
         B, 3, H0, W0, 64, 7, 7, 2, 3, 1);
    {
        size_t tot = (size_t)B * 64 * 64 * 64;
        maxpool_k<<<(unsigned)((tot + 255) / 256), 256, 0, stream>>>(
            stemo, pool, B, 64, 128, 128, 64, 64);
    }

    struct LayerSpec { int cin, wmid, cout, n; };
    const LayerSpec ls[4] = {{64, 64, 256, 3}, {256, 128, 512, 4},
                             {512, 256, 1024, 6}, {1024, 512, 2048, 3}};
    float* xfeat[4] = {xf1, xf2, xf3, xf4};

    const float* curp = pool;
    int curC = 64, curH = 64, curW = 64;
    int base = 21;
    for (int li = 0; li < 4; ++li) {
        int stride0 = (li == 0) ? 1 : 2;
        for (int bi = 0; bi < ls[li].n; ++bi) {
            bool dn = (bi == 0);
            int s = dn ? stride0 : 1;
            int OH = curH / s, OW = curW / s;
            int w1, w2, w3, g1, g2, g3, b1, b2, b3, dwI = -1, dgI = -1, dbI = -1;
            if (dn) {   // sorted: b1,b2,b3,db,dg,dw,g1,g2,g3,w1,w2,w3
                b1 = base + 0; b2 = base + 1; b3 = base + 2;
                dbI = base + 3; dgI = base + 4; dwI = base + 5;
                g1 = base + 6; g2 = base + 7; g3 = base + 8;
                w1 = base + 9; w2 = base + 10; w3 = base + 11; base += 12;
            } else {    // sorted: b1,b2,b3,g1,g2,g3,w1,w2,w3
                b1 = base + 0; b2 = base + 1; b3 = base + 2;
                g1 = base + 3; g2 = base + 4; g3 = base + 5;
                w1 = base + 6; w2 = base + 7; w3 = base + 8; base += 9;
            }
            int wmid = ls[li].wmid, cout = ls[li].cout;
            conv(curp, F(w1), nullptr, F(g1), F(b1), nullptr, T1,
                 B, curC, curH, curW, wmid, 1, 1, 1, 0, 1);
            conv(T1, F(w2), nullptr, F(g2), F(b2), nullptr, T2,
                 B, wmid, curH, curW, wmid, 3, 3, s, 1, 1);
            const float* sc;
            if (dn) {
                conv(curp, F(dwI), nullptr, F(dgI), F(dbI), nullptr, SC,
                     B, curC, curH, curW, cout, 1, 1, s, 0, 0);
                sc = SC;
            } else {
                sc = curp;
            }
            float* dst = (bi == ls[li].n - 1) ? xfeat[li]
                                              : ((curp == bufA) ? bufB : bufA);
            conv(T2, F(w3), nullptr, F(g3), F(b3), sc, dst,
                 B, wmid, OH, OW, cout, 1, 1, 1, 0, 1);
            curp = dst; curC = cout; curH = OH; curW = OW;
        }
    }

    // ================= U-Net decoder =================
    auto cat2 = [&](const float* a, int Ca, const float* b2p, int Cb, int hw) {
        size_t na = (size_t)B * Ca * hw, nb = (size_t)B * Cb * hw;
        copy_ch_k<<<(unsigned)((na + 255) / 256), 256, 0, stream>>>(a, CAT, B, Ca, hw, Ca + Cb, 0);
        copy_ch_k<<<(unsigned)((nb + 255) / 256), 256, 0, stream>>>(b2p, CAT, B, Cb, hw, Ca + Cb, Ca);
    };

    deconv(xf4, F(20), F(19), D1, B, 2048, 8, 8, 1024);          // -> 1024 x 16^2
    cat2(D1, 1024, xf3, 1024, 16 * 16);
    conv(CAT, F(11), F(9), nullptr, nullptr, nullptr, D2, B, 2048, 16, 16, 1024, 3, 3, 1, 1, 1);
    conv(D2,  F(12), F(10), nullptr, nullptr, nullptr, D1, B, 1024, 16, 16, 1024, 3, 3, 1, 1, 1);

    deconv(D1, F(18), F(17), D2, B, 1024, 16, 16, 512);          // -> 512 x 32^2
    cat2(D2, 512, xf2, 512, 32 * 32);
    conv(CAT, F(7), F(5), nullptr, nullptr, nullptr, D1, B, 1024, 32, 32, 512, 3, 3, 1, 1, 1);
    conv(D1,  F(8), F(6), nullptr, nullptr, nullptr, D2, B, 512, 32, 32, 512, 3, 3, 1, 1, 1);

    deconv(D2, F(16), F(15), D1, B, 512, 32, 32, 256);           // -> 256 x 64^2
    cat2(D1, 256, xf1, 256, 64 * 64);
    conv(CAT, F(3), F(1), nullptr, nullptr, nullptr, D2, B, 512, 64, 64, 256, 3, 3, 1, 1, 1);
    conv(D2,  F(4), F(2), nullptr, nullptr, nullptr, D1, B, 256, 64, 64, 256, 3, 3, 1, 1, 1);

    conv(D1, F(14), F(13), nullptr, nullptr, nullptr, (float*)d_out,
         B, 256, 64, 64, 1, 1, 1, 1, 0, 0);

    (void)n_in; (void)out_size; (void)ws_size;
}
